// ChartCover_19885698580847
// MI455X (gfx1250) — matrix operand order repaired and checked by reading, not compiled, not run on hardware
//
#include <hip/hip_runtime.h>
#include <hip/hip_bf16.h>

typedef __attribute__((ext_vector_type(16))) __bf16 v16bf;
typedef __attribute__((ext_vector_type(8)))  float  v8f;

#define CC_M 256
#define CC_D 512
#define CC_B 131072
#define CC_R 32.0f
#define CC_TAU 0.01f
#define CC_EPS 1e-6f

// Tensor Data Mover path; builtin arity differs across toolchains:
//   ROCm 7.2 (clang-22): 5 args;  amdgpu-toolchain (clang-23): 6 args.
#if defined(__has_builtin)
#  if __has_builtin(__builtin_amdgcn_tensor_load_to_lds) && __has_builtin(__builtin_amdgcn_s_wait_tensorcnt)
#    define CC_TDM 1
#  endif
#endif
#ifndef CC_TDM
#  define CC_TDM 0
#endif

// ---------------------------------------------------------------------------
// zero workspace (counts + sums)
// ---------------------------------------------------------------------------
__global__ void cc_zero_kernel(float* p, int n) {
    int i = blockIdx.x * blockDim.x + threadIdx.x;
    if (i < n) p[i] = 0.0f;
}

// ---------------------------------------------------------------------------
// prep: centers f32 -> bf16 copy, and cn[m] = ||centers[m]||^2 (exact f32)
// ---------------------------------------------------------------------------
__global__ __launch_bounds__(256) void cc_prep_kernel(const float* __restrict__ centers,
                                                      unsigned short* __restrict__ c_bf,
                                                      float* __restrict__ cn) {
    __shared__ float red[256];
    const int m = blockIdx.x;
    const int t = threadIdx.x;
    float s = 0.0f;
    for (int k = t; k < CC_D; k += 256) {
        float v = centers[m * CC_D + k];
        __bf16 b = (__bf16)v;
        c_bf[m * CC_D + k] = __builtin_bit_cast(unsigned short, b);
        s += v * v;
    }
    red[t] = s;
    __syncthreads();
    for (int st = 128; st > 0; st >>= 1) {
        if (t < st) red[t] += red[t + st];
        __syncthreads();
    }
    if (t == 0) cn[m] = red[0];
}

// ---------------------------------------------------------------------------
// whiten helper: 4 elements -> bf16 fragment slots [base..base+3], accumulate zn
// ---------------------------------------------------------------------------
__device__ __forceinline__ void cc_whiten4(const float4 a, const float4 m, const float4 s,
                                           float& zn, v16bf& fa, int base) {
    float w0 = (a.x - m.x) * s.x;
    float w1 = (a.y - m.y) * s.y;
    float w2 = (a.z - m.z) * s.z;
    float w3 = (a.w - m.w) * s.w;
    zn += w0 * w0 + w1 * w1 + w2 * w2 + w3 * w3;
    fa[base + 0] = (__bf16)w0;
    fa[base + 1] = (__bf16)w1;
    fa[base + 2] = (__bf16)w2;
    fa[base + 3] = (__bf16)w3;
}

union cc_bpack { uint4 u[2]; v16bf v; };

#if CC_TDM
typedef unsigned int cc_u32x4 __attribute__((ext_vector_type(4)));
typedef int          cc_i32x4 __attribute__((ext_vector_type(4)));
typedef int          cc_i32x8 __attribute__((ext_vector_type(8)));

// Issue one TDM 2D tile load: 256 rows x 64 bytes from c_bf (row stride 1024 B,
// tile start at byte column kt*64) into contiguous LDS at lds_byte.
// D# bitfields per CDNA5 ISA §8.3/8.4; data_size = 8 bytes.
__device__ __forceinline__ void cc_tdm_issue(const unsigned short* c_bf,
                                             unsigned lds_byte, int kt) {
    unsigned long long ga =
        (unsigned long long)(const void*)c_bf + (unsigned long long)kt * 64ull;
    cc_u32x4 g0;
    g0[0] = 1u;                                            // count=1, user D#
    g0[1] = lds_byte;                                      // lds_addr [63:32]
    g0[2] = (unsigned)(ga & 0xffffffffull);                // global_addr lo
    g0[3] = (unsigned)((ga >> 32) & 0x01ffffffull)         // global_addr hi
          | 0x80000000u;                                   // type=2 ("image")
    cc_i32x8 g1;
    g1[0] = (3 << 16);          // workgroup_mask=0, data_size=3 (8B units)
    g1[1] = (int)(128u << 16);  // tensor_dim0 = 128 (row = 1024B = 128*8B)
    g1[2] = (int)(256u << 16);  // tensor_dim1 = 256 rows
    g1[3] = (int)(8u << 16);    // tile_dim0 = 8 (64B per row)
    g1[4] = 256;                // tile_dim1 = 256 rows, tile_dim2 = 0
    g1[5] = 128;                // tensor_dim0_stride = 128 (8B units)
    g1[6] = 0;
    g1[7] = 0;
    cc_i32x4 gz4 = {0, 0, 0, 0};                 // 2D tile: groups 2/3 unused
#if __clang_major__ >= 23
    cc_i32x8 gz8 = {0, 0, 0, 0, 0, 0, 0, 0};
    __builtin_amdgcn_tensor_load_to_lds(g0, g1, gz4, gz4, gz8, 0);
#else
    __builtin_amdgcn_tensor_load_to_lds(g0, g1, gz4, gz4, 0);
#endif
}
#endif

// ---------------------------------------------------------------------------
// main kernel: fused whiten + cdist (bf16 WMMA) + argmin + masks + scatter-add
// block = 256 threads (8 waves); each wave: 16 batch rows x all 256 centers
// grid.x = B / 128 = 1024
// ---------------------------------------------------------------------------
__global__ __launch_bounds__(256) void cc_dist_kernel(
    const float* __restrict__ z,
    const float* __restrict__ stats_mean,
    const float* __restrict__ stats_var,
    const unsigned short* __restrict__ c_bf,
    const float* __restrict__ cn,
    float* __restrict__ sums,        // [M, D] atomic accumulation
    float* __restrict__ counts,      // [M]
    float* __restrict__ out_dists,   // [B, M]
    float* __restrict__ out_hidx,    // [B]  (index as float)
    float* __restrict__ out_masks)   // [B, M] (0/1 float)
{
#if CC_TDM
    __shared__ unsigned short cs[2][CC_M * 32];  // 2 x 16 KB double-buffered slab
#else
    __shared__ unsigned short cs[1][CC_M * 32];  // 16 KB K-slab of all centers
#endif
    __shared__ float mean_s[CC_D];
    __shared__ float inv_s[CC_D];
    __shared__ float cn_s[CC_M];
    __shared__ float zn_s[8][16];
    __shared__ int   hi_s[8][16];

    const int tid  = threadIdx.x;
    const int w    = tid >> 5;
    const int lane = tid & 31;
    const int half = lane >> 4;     // 0: lanes 0-15, 1: lanes 16-31
    const int l16  = lane & 15;

    // stage whitening params + center norms
    mean_s[tid]       = stats_mean[tid];
    mean_s[tid + 256] = stats_mean[tid + 256];
    inv_s[tid]        = rsqrtf(stats_var[tid] + CC_EPS);
    inv_s[tid + 256]  = rsqrtf(stats_var[tid + 256] + CC_EPS);
    cn_s[tid]         = cn[tid];
    __syncthreads();

    const int rowBase = blockIdx.x * 128 + w * 16;   // this wave's 16 rows
    const float* zrow = z + (size_t)(rowBase + l16) * CC_D;

    v8f acc[16];
#pragma unroll
    for (int nt = 0; nt < 16; ++nt) acc[nt] = (v8f){0, 0, 0, 0, 0, 0, 0, 0};

    float znp = 0.0f;

#if CC_TDM
    unsigned ldsA = (unsigned)(unsigned long long)
        (__attribute__((address_space(3))) unsigned short*)(&cs[0][0]);
    unsigned ldsB = (unsigned)(unsigned long long)
        (__attribute__((address_space(3))) unsigned short*)(&cs[1][0]);
    if (w == 0) cc_tdm_issue(c_bf, ldsA, 0);     // prefetch slab 0
#endif

    for (int kt = 0; kt < CC_D / 32; ++kt) {
        const int k0   = kt * 32;
        const int koff = k0 + half * 8;   // this half-wave's K sub-block start

        // ---- A fragment: load 16 f32 of z, whiten, pack to bf16 ISA layout
        float4 a0 = *(const float4*)(zrow + koff);
        float4 a1 = *(const float4*)(zrow + koff + 4);
        float4 a2 = *(const float4*)(zrow + koff + 16);
        float4 a3 = *(const float4*)(zrow + koff + 20);
        float4 m0 = *(const float4*)(mean_s + koff);
        float4 m1 = *(const float4*)(mean_s + koff + 4);
        float4 m2 = *(const float4*)(mean_s + koff + 16);
        float4 m3 = *(const float4*)(mean_s + koff + 20);
        float4 s0 = *(const float4*)(inv_s + koff);
        float4 s1 = *(const float4*)(inv_s + koff + 4);
        float4 s2 = *(const float4*)(inv_s + koff + 16);
        float4 s3 = *(const float4*)(inv_s + koff + 20);

        v16bf fa;
        cc_whiten4(a0, m0, s0, znp, fa, 0);
        cc_whiten4(a1, m1, s1, znp, fa, 4);
        cc_whiten4(a2, m2, s2, znp, fa, 8);
        cc_whiten4(a3, m3, s3, znp, fa, 12);

#if CC_TDM
        // ---- double-buffered TDM staging: wait current slab, prefetch next
        if (w == 0) __builtin_amdgcn_s_wait_tensorcnt(0);
        __syncthreads();   // slab[kt&1] visible to all; slab[kt&1 ^ 1] free
        if (kt + 1 < CC_D / 32 && w == 0)
            cc_tdm_issue(c_bf, (kt & 1) ? ldsA : ldsB, kt + 1);
        const unsigned short* csb = cs[kt & 1];
#else
        // ---- manual staging: thread t copies center t's 64 bytes
        __syncthreads();   // previous slab fully consumed
        {
            const uint4* src = (const uint4*)(c_bf + (size_t)tid * CC_D + k0);
            uint4* dst = (uint4*)(cs[0] + tid * 32);
            dst[0] = src[0]; dst[1] = src[1]; dst[2] = src[2]; dst[3] = src[3];
        }
        __syncthreads();
        const unsigned short* csb = cs[0];
#endif

        // ---- 16 WMMAs: all 256 centers for this K chunk
#pragma unroll
        for (int nt = 0; nt < 16; ++nt) {
            const int c = nt * 16 + l16;          // this lane's center (B column)
            cc_bpack bp;
            const uint4* bsrc = (const uint4*)(csb + c * 32 + half * 8);
            bp.u[0] = bsrc[0];
            bp.u[1] = bsrc[2];
            acc[nt] = __builtin_amdgcn_wmma_f32_16x16x32_bf16(
                false, fa, false, bp.v, (short)0, acc[nt], false, false);
        }
    }

    // ---- complete ||z_w||^2 per row (lane L and L+16 hold complementary halves)
    float znf = znp + __shfl_xor(znp, 16);
    zn_s[w][l16] = znf;

    // ---- epilogue: d2 -> dist, masks, per-row argmin
    float mv[8];
    int   mi[8];
#pragma unroll
    for (int j = 0; j < 8; ++j) { mv[j] = 3.4e38f; mi[j] = 0; }

#pragma unroll
    for (int nt = 0; nt < 16; ++nt) {
        const int col = nt * 16 + l16;
        const float cnv = cn_s[col];
#pragma unroll
        for (int j = 0; j < 8; ++j) {
            const int rl   = j + half * 8;            // local row 0..15
            const int rowG = rowBase + rl;
            float d2   = zn_s[w][rl] + cnv - 2.0f * acc[nt][j];
            float dist = sqrtf(fmaxf(d2, 0.0f));
            out_dists[(size_t)rowG * CC_M + col] = dist;
            out_masks[(size_t)rowG * CC_M + col] = (dist <= CC_R) ? 1.0f : 0.0f;
            if (dist < mv[j]) { mv[j] = dist; mi[j] = col; }
        }
    }

    // ---- cross-lane min reduce within each 16-lane half (same row set)
#pragma unroll
    for (int j = 0; j < 8; ++j) {
#pragma unroll
        for (int msk = 1; msk < 16; msk <<= 1) {
            float ov = __shfl_xor(mv[j], msk);
            int   oi = __shfl_xor(mi[j], msk);
            if (ov < mv[j] || (ov == mv[j] && oi < mi[j])) { mv[j] = ov; mi[j] = oi; }
        }
    }

    if (l16 == 0) {
#pragma unroll
        for (int j = 0; j < 8; ++j) {
            const int rl = j + half * 8;
            out_hidx[rowBase + rl] = (float)mi[j];
            hi_s[w][rl] = mi[j];
            atomicAdd(&counts[mi[j]], 1.0f);
        }
    }

    // ---- fused scatter-add of whitened rows into sums[hard_idx] (L2-resident)
    for (int r = 0; r < 16; ++r) {
        const int c = hi_s[w][r];
        const float* zr = z + (size_t)(rowBase + r) * CC_D;
        float* srow = sums + (size_t)c * CC_D;
        for (int k = lane; k < CC_D; k += 32) {
            float zw = (zr[k] - mean_s[k]) * inv_s[k];
            atomicAdd(&srow[k], zw);
        }
    }
}

// ---------------------------------------------------------------------------
// finalize: EMA center update
// ---------------------------------------------------------------------------
__global__ __launch_bounds__(256) void cc_finalize_kernel(const float* __restrict__ centers,
                                                          const float* __restrict__ sums,
                                                          const float* __restrict__ counts,
                                                          float* __restrict__ out_nc) {
    int i = blockIdx.x * blockDim.x + threadIdx.x;   // < M*D
    int m = i >> 9;
    float cnt = counts[m];
    float c   = centers[i];
    float nc  = c;
    if (cnt > 0.0f) nc = (1.0f - CC_TAU) * c + CC_TAU * (sums[i] / fmaxf(cnt, 1.0f));
    out_nc[i] = nc;
}

// ---------------------------------------------------------------------------
extern "C" void kernel_launch(void* const* d_in, const int* in_sizes, int n_in,
                              void* d_out, int out_size, void* d_ws, size_t ws_size,
                              hipStream_t stream) {
    const float* z          = (const float*)d_in[0];   // [B, D]
    const float* centers    = (const float*)d_in[1];   // [M, D]
    const float* stats_mean = (const float*)d_in[2];   // [D]
    const float* stats_var  = (const float*)d_in[3];   // [D]

    // workspace layout
    float* cn      = (float*)d_ws;                         // [M]
    float* counts  = cn + CC_M;                            // [M]
    float* sums    = counts + CC_M;                        // [M, D]
    unsigned short* c_bf = (unsigned short*)(sums + CC_M * CC_D);  // [M, D] bf16

    // output layout (flat f32, in reference return order)
    float* out_dists = (float*)d_out;                      // [B, M]
    float* out_hidx  = out_dists + (size_t)CC_B * CC_M;    // [B]
    float* out_masks = out_hidx + CC_B;                    // [B, M]
    float* out_nc    = out_masks + (size_t)CC_B * CC_M;    // [M, D]

    // 1) zero counts + sums (contiguous)
    {
        int n = CC_M + CC_M * CC_D;
        cc_zero_kernel<<<(n + 255) / 256, 256, 0, stream>>>(counts, n);
    }
    // 2) centers -> bf16 + norms
    cc_prep_kernel<<<CC_M, 256, 0, stream>>>(centers, c_bf, cn);
    // 3) fused whiten + WMMA cdist + argmin + masks + scatter
    cc_dist_kernel<<<CC_B / 128, 256, 0, stream>>>(z, stats_mean, stats_var, c_bf, cn,
                                                   sums, counts,
                                                   out_dists, out_hidx, out_masks);
    // 4) EMA update
    cc_finalize_kernel<<<(CC_M * CC_D) / 256, 256, 0, stream>>>(centers, sums, counts, out_nc);
}
